// RFMBlock_85847806312925
// MI455X (gfx1250) — matrix operand order repaired and checked by reading, not compiled, not run on hardware
//
#include <hip/hip_runtime.h>
#include <hip/hip_fp16.h>

typedef __attribute__((ext_vector_type(16))) _Float16 v16h;
typedef __attribute__((ext_vector_type(8)))  _Float16 v8h;
typedef __attribute__((ext_vector_type(4)))  _Float16 v4h;
typedef __attribute__((ext_vector_type(8)))  float    v8f;

#define DEV __device__ __forceinline__

// ---------------------------------------------------------------------------
// WMMA helpers (CDNA5 16x16x32 f16 -> f32)
// ---------------------------------------------------------------------------
DEV v8f wmma_f16(v16h a, v16h b, v8f c) {
  // 8 args: (neg_a, A, neg_b, B, c_mod, C, reuse_a, reuse_b)
  return __builtin_amdgcn_wmma_f32_16x16x32_f16(false, a, false, b, (short)0, c,
                                                false, false);
}

// A fragment 16x32 from LDS row-major X[ldx] (rows m0..m0+15, K k0..k0+31).
// Contiguous K order (this code's choice, same as B): lanes 0-15 = M, lane-group selects K sub-ranges:
//   halves[0..7]  = K  (k0 + 16*hi) .. +7
//   halves[8..15] = K  (k0 + 16*hi + 8) .. +7
DEV v16h load_afrag(const _Float16* X, int ldx, int m0, int k0, int lane) {
  const int r15 = lane & 15, hi = lane >> 4;
  const _Float16* p = X + (m0 + r15) * ldx + k0 + 16 * hi;
  union { v16h v; v8h h[2]; } a;
  a.h[0] = *(const v8h*)p;
  a.h[1] = *(const v8h*)(p + 8);
  return a.v;
}

// B fragment 32x16 from TRANSPOSED weights Wt[Nout][K] (cols n0..n0+15).
// Contiguous K order (this code's choice, same as A): lane = N column, halves = 16 consecutive K (lane-group picks
// K 0..15 vs 16..31) -> one contiguous 32B read per lane.
DEV v16h load_bfrag(const _Float16* Wt, int K, int n0, int k0, int lane) {
  const int r15 = lane & 15, hi = lane >> 4;
  const _Float16* p = Wt + (size_t)(n0 + r15) * K + k0 + 16 * hi;
  return *(const v16h*)p;
}

template <int KTOT, int NMT>
DEV void gemm_stage(const _Float16* X, int ldx, const _Float16* Wt,
                    int n0, int mt0, int lane, v8f acc[NMT]) {
#pragma unroll
  for (int k0 = 0; k0 < KTOT; k0 += 32) {
    v16h b = load_bfrag(Wt, KTOT, n0, k0, lane);
#pragma unroll
    for (int t = 0; t < NMT; ++t) {
      v16h a = load_afrag(X, ldx, (mt0 + t) * 16, k0, lane);
      acc[t] = wmma_f16(a, b, acc[t]);
    }
  }
}

// C/D layout: VGPR r holds row (m0 + r + 8*hi), col = n0 + (lane&15).
template <bool RELU>
DEV void store_hidden(const v8f& c, const float* bias, _Float16* H, int ldh,
                      int m0, int n0, int lane) {
  const int r15 = lane & 15, hi = lane >> 4;
  const float bv = bias[n0 + r15];
#pragma unroll
  for (int r = 0; r < 8; ++r) {
    float v = c[r] + bv;
    if (RELU) v = fmaxf(v, 0.0f);
    H[(m0 + r + 8 * hi) * ldh + n0 + r15] = (_Float16)v;
  }
}

// ---------------------------------------------------------------------------
// Weight f32 -> f16 transpose:  W[K][Nout]  ->  Wt[Nout][K]
// ---------------------------------------------------------------------------
__global__ __launch_bounds__(256) void convw_kernel(const float* __restrict__ W,
                                                    _Float16* __restrict__ Wt,
                                                    int K, int Nout) {
  int i = blockIdx.x * 256 + threadIdx.x;
  if (i < K * Nout) {
    int k = i / Nout, n = i - k * Nout;
    Wt[(size_t)n * K + k] = (_Float16)W[i];
  }
}

// spread[64][16*64] -> comb[16*64]
__global__ __launch_bounds__(256) void reduce_spread_kernel(
    const float* __restrict__ spread, float* __restrict__ comb) {
  int i = blockIdx.x * 256 + threadIdx.x;
  if (i < 16 * 64) {
    float s = 0.0f;
    for (int k = 0; k < 64; ++k) s += spread[k * 1024 + i];
    comb[i] = s;
  }
}

// ---------------------------------------------------------------------------
// Edge MLP: [E,256] -> 128 -> 128 -> 64, + scatter to msgs and per-graph sums
// ---------------------------------------------------------------------------
__global__ __launch_bounds__(256) void edge_mlp_kernel(
    const float* __restrict__ edge_feat, const float* __restrict__ node_feat,
    const float* __restrict__ g_repr, const int* __restrict__ src,
    const int* __restrict__ dst, const int* __restrict__ e2g,
    const _Float16* __restrict__ W1t, const float* __restrict__ b1,
    const _Float16* __restrict__ W2t, const float* __restrict__ b2,
    const _Float16* __restrict__ W3t, const float* __restrict__ b3,
    float* __restrict__ e_out, float* __restrict__ msgs,
    float* __restrict__ e_spread, int E) {
  __shared__ _Float16 X[64 * 256];
  __shared__ _Float16 H1[64 * 128];
  __shared__ _Float16 H2[64 * 128];
  __shared__ float accG[16 * 64];
  __shared__ int sDst[64];
  __shared__ int sGid[64];

  const int tid = threadIdx.x;
  const int lane = tid & 31;
  const int wave = tid >> 5;
  const int e0 = blockIdx.x * 64;

  for (int i = tid; i < 16 * 64; i += 256) accG[i] = 0.0f;
  if (tid < 64) {
    int e = min(e0 + tid, E - 1);
    sDst[tid] = dst[e];
    sGid[tid] = e2g[e];
  }

  {  // gather + f32->f16: row = edge_feat | node[src] | node[dst] | g_repr
    int row = tid >> 2, q = tid & 3;
    int e = min(e0 + row, E - 1);
    const float* sp;
    if (q == 0)      sp = edge_feat + (size_t)e * 64;
    else if (q == 1) sp = node_feat + (size_t)src[e] * 64;
    else if (q == 2) sp = node_feat + (size_t)dst[e] * 64;
    else             sp = g_repr + (size_t)e2g[e] * 64;
    _Float16* dp = X + row * 256 + q * 64;
#pragma unroll
    for (int i = 0; i < 64; i += 4) {
      float4 f = *(const float4*)(sp + i);
      v4h h = {(_Float16)f.x, (_Float16)f.y, (_Float16)f.z, (_Float16)f.w};
      *(v4h*)(dp + i) = h;
    }
  }
  __syncthreads();

  {  // stage 1: [64,256] @ W1 + relu
    v8f c[4] = {};
    const int n0 = wave * 16;
    gemm_stage<256, 4>(X, 256, W1t, n0, 0, lane, c);
#pragma unroll
    for (int t = 0; t < 4; ++t) store_hidden<true>(c[t], b1, H1, 128, t * 16, n0, lane);
  }
  __syncthreads();

  {  // stage 2: [64,128] @ W2 + relu
    v8f c[4] = {};
    const int n0 = wave * 16;
    gemm_stage<128, 4>(H1, 128, W2t, n0, 0, lane, c);
#pragma unroll
    for (int t = 0; t < 4; ++t) store_hidden<true>(c[t], b2, H2, 128, t * 16, n0, lane);
  }
  __syncthreads();

  {  // stage 3: [64,128] @ W3 -> e_out, scatter-add
    v8f c[2] = {};
    const int n0 = (wave & 3) * 16;
    const int mt0 = (wave >> 2) * 2;
    gemm_stage<128, 2>(H2, 128, W3t, n0, mt0, lane, c);
    const int r15 = lane & 15, hi = lane >> 4;
    const float bv = b3[n0 + r15];
    const int col = n0 + r15;
#pragma unroll
    for (int t = 0; t < 2; ++t) {
#pragma unroll
      for (int r = 0; r < 8; ++r) {
        int lrow = (mt0 + t) * 16 + r + 8 * hi;
        int e = e0 + lrow;
        if (e < E) {
          float v = c[t][r] + bv;
          e_out[(size_t)e * 64 + col] = v;
          atomicAdd(&msgs[(size_t)sDst[lrow] * 64 + col], v);
          atomicAdd(&accG[sGid[lrow] * 64 + col], v);
        }
      }
    }
  }
  __syncthreads();

  {  // flush per-graph partials to one of 64 spread slots (contention /64)
    float* dp = e_spread + (size_t)(blockIdx.x & 63) * 1024;
    for (int i = tid; i < 1024; i += 256) atomicAdd(&dp[i], accG[i]);
  }
}

// ---------------------------------------------------------------------------
// Node MLP: [N,192] -> 128 -> 128 -> 64, + per-graph sums
// ---------------------------------------------------------------------------
__global__ __launch_bounds__(256) void node_mlp_kernel(
    const float* __restrict__ node_feat, const float* __restrict__ msgs,
    const float* __restrict__ g_repr, const int* __restrict__ n2g,
    const _Float16* __restrict__ W1t, const float* __restrict__ b1,
    const _Float16* __restrict__ W2t, const float* __restrict__ b2,
    const _Float16* __restrict__ W3t, const float* __restrict__ b3,
    float* __restrict__ n_out, float* __restrict__ n_spread, int N) {
  __shared__ _Float16 X[64 * 192];
  __shared__ _Float16 H1[64 * 128];
  __shared__ _Float16 H2[64 * 128];
  __shared__ float accG[16 * 64];
  __shared__ int sGid[64];

  const int tid = threadIdx.x;
  const int lane = tid & 31;
  const int wave = tid >> 5;
  const int n0r = blockIdx.x * 64;

  for (int i = tid; i < 16 * 64; i += 256) accG[i] = 0.0f;
  if (tid < 64) sGid[tid] = n2g[min(n0r + tid, N - 1)];

  for (int idx = tid; idx < 64 * 3; idx += 256) {
    int row = idx / 3, q = idx - row * 3;
    int n = min(n0r + row, N - 1);
    const float* sp;
    if (q == 0)      sp = node_feat + (size_t)n * 64;
    else if (q == 1) sp = msgs + (size_t)n * 64;
    else             sp = g_repr + (size_t)n2g[n] * 64;
    _Float16* dp = X + row * 192 + q * 64;
#pragma unroll
    for (int i = 0; i < 64; i += 4) {
      float4 f = *(const float4*)(sp + i);
      v4h h = {(_Float16)f.x, (_Float16)f.y, (_Float16)f.z, (_Float16)f.w};
      *(v4h*)(dp + i) = h;
    }
  }
  __syncthreads();

  {
    v8f c[4] = {};
    const int n0 = wave * 16;
    gemm_stage<192, 4>(X, 192, W1t, n0, 0, lane, c);
#pragma unroll
    for (int t = 0; t < 4; ++t) store_hidden<true>(c[t], b1, H1, 128, t * 16, n0, lane);
  }
  __syncthreads();
  {
    v8f c[4] = {};
    const int n0 = wave * 16;
    gemm_stage<128, 4>(H1, 128, W2t, n0, 0, lane, c);
#pragma unroll
    for (int t = 0; t < 4; ++t) store_hidden<true>(c[t], b2, H2, 128, t * 16, n0, lane);
  }
  __syncthreads();
  {
    v8f c[2] = {};
    const int n0 = (wave & 3) * 16;
    const int mt0 = (wave >> 2) * 2;
    gemm_stage<128, 2>(H2, 128, W3t, n0, mt0, lane, c);
    const int r15 = lane & 15, hi = lane >> 4;
    const float bv = b3[n0 + r15];
    const int col = n0 + r15;
#pragma unroll
    for (int t = 0; t < 2; ++t) {
#pragma unroll
      for (int r = 0; r < 8; ++r) {
        int lrow = (mt0 + t) * 16 + r + 8 * hi;
        int n = n0r + lrow;
        if (n < N) {
          float v = c[t][r] + bv;
          n_out[(size_t)n * 64 + col] = v;
          atomicAdd(&accG[sGid[lrow] * 64 + col], v);
        }
      }
    }
  }
  __syncthreads();
  {
    float* dp = n_spread + (size_t)(blockIdx.x & 63) * 1024;
    for (int i = tid; i < 1024; i += 256) atomicAdd(&dp[i], accG[i]);
  }
}

// ---------------------------------------------------------------------------
// Global MLP: [16,192] -> 128 -> 128 -> 64
// ---------------------------------------------------------------------------
__global__ __launch_bounds__(256) void global_mlp_kernel(
    const float* __restrict__ n_comb, const float* __restrict__ e_comb,
    const float* __restrict__ g_repr,
    const _Float16* __restrict__ W1t, const float* __restrict__ b1,
    const _Float16* __restrict__ W2t, const float* __restrict__ b2,
    const _Float16* __restrict__ W3t, const float* __restrict__ b3,
    float* __restrict__ u_out) {
  __shared__ _Float16 X[16 * 192];
  __shared__ _Float16 H1[16 * 128];
  __shared__ _Float16 H2[16 * 128];
  const int tid = threadIdx.x, lane = tid & 31, wave = tid >> 5;

  for (int idx = tid; idx < 16 * 3; idx += 256) {
    int row = idx / 3, q = idx - row * 3;
    const float* sp = (q == 0) ? n_comb + row * 64
                    : (q == 1) ? e_comb + row * 64
                               : g_repr + row * 64;
    _Float16* dp = X + row * 192 + q * 64;
#pragma unroll
    for (int i = 0; i < 64; i += 4) {
      float4 f = *(const float4*)(sp + i);
      v4h h = {(_Float16)f.x, (_Float16)f.y, (_Float16)f.z, (_Float16)f.w};
      *(v4h*)(dp + i) = h;
    }
  }
  __syncthreads();
  {
    v8f c[1] = {};
    const int n0 = wave * 16;
    gemm_stage<192, 1>(X, 192, W1t, n0, 0, lane, c);
    store_hidden<true>(c[0], b1, H1, 128, 0, n0, lane);
  }
  __syncthreads();
  {
    v8f c[1] = {};
    const int n0 = wave * 16;
    gemm_stage<128, 1>(H1, 128, W2t, n0, 0, lane, c);
    store_hidden<true>(c[0], b2, H2, 128, 0, n0, lane);
  }
  __syncthreads();
  if (wave < 4) {  // wave-uniform branch: EXEC all-ones inside
    v8f c[1] = {};
    const int n0 = wave * 16;
    gemm_stage<128, 1>(H2, 128, W3t, n0, 0, lane, c);
    const int r15 = lane & 15, hi = lane >> 4;
    const float bv = b3[n0 + r15];
#pragma unroll
    for (int r = 0; r < 8; ++r)
      u_out[(size_t)(r + 8 * hi) * 64 + n0 + r15] = c[0][r] + bv;
  }
}

// ---------------------------------------------------------------------------
// Host: orchestrate the pipeline (graph-capture safe)
// ---------------------------------------------------------------------------
extern "C" void kernel_launch(void* const* d_in, const int* in_sizes, int n_in,
                              void* d_out, int out_size, void* d_ws, size_t ws_size,
                              hipStream_t stream) {
  const float* edge_feat = (const float*)d_in[0];
  const float* node_feat = (const float*)d_in[1];
  const float* g_repr    = (const float*)d_in[2];
  const int* src = (const int*)d_in[3];
  const int* dst = (const int*)d_in[4];
  const int* e2g = (const int*)d_in[5];
  const int* n2g = (const int*)d_in[6];
  const float* We1 = (const float*)d_in[7];  const float* be1 = (const float*)d_in[8];
  const float* We2 = (const float*)d_in[9];  const float* be2 = (const float*)d_in[10];
  const float* We3 = (const float*)d_in[11]; const float* be3 = (const float*)d_in[12];
  const float* Wn1 = (const float*)d_in[13]; const float* bn1 = (const float*)d_in[14];
  const float* Wn2 = (const float*)d_in[15]; const float* bn2 = (const float*)d_in[16];
  const float* Wn3 = (const float*)d_in[17]; const float* bn3 = (const float*)d_in[18];
  const float* Wu1 = (const float*)d_in[19]; const float* bu1 = (const float*)d_in[20];
  const float* Wu2 = (const float*)d_in[21]; const float* bu2 = (const float*)d_in[22];
  const float* Wu3 = (const float*)d_in[23]; const float* bu3 = (const float*)d_in[24];

  const int E = in_sizes[3];      // 400000
  const int N = in_sizes[6];      // 50000

  char* ws = (char*)d_ws;
  size_t off = 0;
  auto alloc = [&](size_t bytes) {
    void* p = ws + off;
    off += (bytes + 511) & ~(size_t)511;
    return p;
  };
  _Float16* We1t = (_Float16*)alloc(256 * 128 * 2);
  _Float16* We2t = (_Float16*)alloc(128 * 128 * 2);
  _Float16* We3t = (_Float16*)alloc(128 * 64 * 2);
  _Float16* Wn1t = (_Float16*)alloc(192 * 128 * 2);
  _Float16* Wn2t = (_Float16*)alloc(128 * 128 * 2);
  _Float16* Wn3t = (_Float16*)alloc(128 * 64 * 2);
  _Float16* Wu1t = (_Float16*)alloc(192 * 128 * 2);
  _Float16* Wu2t = (_Float16*)alloc(128 * 128 * 2);
  _Float16* Wu3t = (_Float16*)alloc(128 * 64 * 2);
  float* msgs     = (float*)alloc((size_t)N * 64 * 4);
  float* e_spread = (float*)alloc(64 * 1024 * 4);
  float* n_spread = (float*)alloc(64 * 1024 * 4);
  float* e_comb   = (float*)alloc(16 * 64 * 4);
  float* n_comb   = (float*)alloc(16 * 64 * 4);

  auto conv = [&](const float* W, _Float16* Wt, int K, int Nout) {
    int total = K * Nout;
    convw_kernel<<<(total + 255) / 256, 256, 0, stream>>>(W, Wt, K, Nout);
  };
  conv(We1, We1t, 256, 128); conv(We2, We2t, 128, 128); conv(We3, We3t, 128, 64);
  conv(Wn1, Wn1t, 192, 128); conv(Wn2, Wn2t, 128, 128); conv(Wn3, Wn3t, 128, 64);
  conv(Wu1, Wu1t, 192, 128); conv(Wu2, Wu2t, 128, 128); conv(Wu3, Wu3t, 128, 64);

  hipMemsetAsync(msgs, 0, (size_t)N * 64 * 4, stream);
  hipMemsetAsync(e_spread, 0, 64 * 1024 * 4, stream);
  hipMemsetAsync(n_spread, 0, 64 * 1024 * 4, stream);

  float* e_out = (float*)d_out;
  float* n_out = e_out + (size_t)E * 64;
  float* u_out = n_out + (size_t)N * 64;

  edge_mlp_kernel<<<(E + 63) / 64, 256, 0, stream>>>(
      edge_feat, node_feat, g_repr, src, dst, e2g,
      We1t, be1, We2t, be2, We3t, be3, e_out, msgs, e_spread, E);

  reduce_spread_kernel<<<4, 256, 0, stream>>>(e_spread, e_comb);

  node_mlp_kernel<<<(N + 63) / 64, 256, 0, stream>>>(
      node_feat, msgs, g_repr, n2g,
      Wn1t, bn1, Wn2t, bn2, Wn3t, bn3, n_out, n_spread, N);

  reduce_spread_kernel<<<4, 256, 0, stream>>>(n_spread, n_comb);

  global_mlp_kernel<<<1, 256, 0, stream>>>(
      n_comb, e_comb, g_repr, Wu1t, bu1, Wu2t, bu2, Wu3t, bu3, u_out);

  (void)n_in; (void)out_size; (void)ws_size;
}
